// ConvexReLUCNN_60103772340826
// MI455X (gfx1250) — matrix operand order repaired and checked by reading, not compiled, not run on hardware
//
#include <hip/hip_runtime.h>
#include <hip/hip_bf16.h>

// ---------------------------------------------------------------------------
// ConvexReLUCNN: out[b,o] = sum_{m,k} (sum_l U[b,k,l] * G[m,l]) * (v-w)[m,k,o]
// Stage 1: bf16 WMMA GEMM  (13824 x 3844) x (3844 x 512) -> Abuf (f32)
//          A-operand gathered from x (unfold view) with clamp (no predication;
//          padded K columns of Gt are zero so clamped A values are harmless).
// Stage 2: tiny reduction  Abuf x (v-w) -> out (512 x 10)
// ---------------------------------------------------------------------------

typedef __attribute__((ext_vector_type(16))) __bf16 v16bf;
typedef __attribute__((ext_vector_type(8)))  __bf16 v8bf;
typedef __attribute__((ext_vector_type(8)))  float  v8f;

#define L_DIM   3844          // 62*62 patches
#define LP_DIM  3872          // padded to multiple of 32
#define KC      32            // k-chunk per WMMA step
#define KITERS  (LP_DIM / KC) // 121
#define M_DIM   512
#define R_DIM   13824         // B * 27
#define WO      62

#define LDS_STRIDE 40         // 32 + 8 halves of padding (keeps 16B alignment)

__device__ __forceinline__ unsigned short f32_to_bf16_rne(float f) {
    unsigned int u = __float_as_uint(f);
    unsigned int r = u + 0x7fffu + ((u >> 16) & 1u);
    return (unsigned short)(r >> 16);
}

// --------------------------- prep: G -> Gt (bf16, K-blocked) ----------------
// Gt layout: [lb][m][lo], lb = l/32, lo = l%32. Zero-padded for l >= 3844.
__global__ void prep_gt_kernel(const float* __restrict__ G,
                               unsigned short* __restrict__ Gt) {
    int idx = blockIdx.x * 256 + threadIdx.x;
    if (idx >= KITERS * M_DIM * KC) return;
    int lo = idx & 31;
    int m  = (idx >> 5) & (M_DIM - 1);
    int lb = idx >> 14;                 // /(32*512)
    int l  = lb * KC + lo;
    float g = (l < L_DIM) ? G[m * L_DIM + l] : 0.0f;
    Gt[idx] = f32_to_bf16_rne(g);
}

// --------------------------- stage 1: WMMA GEMM -----------------------------
// Block: 256 threads (8 wave32), 128x256 tile of Abuf. Each wave: 32x128.
// 16 v_wmma per k-step per wave; acc = 2x8 v8f = 128 VGPRs.
__global__ void __launch_bounds__(256)
unfold_gemm_wmma_kernel(const float* __restrict__ x,
                        const unsigned short* __restrict__ Gt,
                        float* __restrict__ Abuf) {
    __shared__ unsigned short AsU[128 * LDS_STRIDE]; // A tile, [row][k]
    __shared__ unsigned short BsU[256 * LDS_STRIDE]; // B tile, [m][k]

    const int tid  = threadIdx.x;
    const int m0   = blockIdx.x * 256;   // 0 or 256
    const int r0   = blockIdx.y * 128;   // 0..13696

    const int lane   = tid & 31;
    const int th     = lane >> 4;        // lane half
    const int tl     = lane & 15;
    const int waveId = tid >> 5;
    const int wr     = waveId & 3;       // row group of 32
    const int wc     = waveId >> 2;      // col group of 128

    // --- per-thread A-tile gather assignment (row fixed across k-iters) ---
    const int rowLocal = tid >> 1;            // 0..127
    const int lo0      = (tid & 1) * 16;      // 0 or 16
    const int r  = r0 + rowLocal;
    const int b  = r / 27;
    const int kk = r - b * 27;
    const int c  = kk / 9;
    const int ij = kk - c * 9;
    const int ii = ij / 3;
    const int jj = ij - ii * 3;
    // x element at patch (oh,ow): xbase[oh*64 + ow]; oh,ow <= 61 always valid
    const float* xbase = x + ((b * 3 + c) * 64 + ii) * 64 + jj;

    // --- per-thread B-tile load assignment: one full 32-k row of Gt ---
    const unsigned short* bsrc = Gt + (size_t)(m0 + tid) * KC;

    v8f acc[2][8] = {};

    union FragBF { v16bf v; v8bf h[2]; };

    for (int it = 0; it < KITERS; ++it) {
        const int l0 = it * KC;

        // ---- load A tile (unfold gather, clamped -> branch-free) ----
        {
            union { unsigned short s[16]; uint4 q[2]; } tmp;
            int l = l0 + lo0;
            #pragma unroll
            for (int q = 0; q < 16; ++q, ++l) {
                int lc = (l < L_DIM) ? l : (L_DIM - 1);  // v_min, no branch
                int oh = lc / WO;
                int ow = lc - oh * WO;
                tmp.s[q] = f32_to_bf16_rne(xbase[oh * 64 + ow]);
            }
            uint4* dst = (uint4*)(AsU + rowLocal * LDS_STRIDE + lo0);
            dst[0] = tmp.q[0];
            dst[1] = tmp.q[1];
        }

        // ---- load B tile: 256 rows x 32 k, 64B per thread, coalesced ----
        {
            const unsigned short* src = bsrc + (size_t)it * (M_DIM * KC);
            const uint4* s4 = (const uint4*)src;
            uint4 d0 = s4[0], d1 = s4[1], d2 = s4[2], d3 = s4[3];
            uint4* dst = (uint4*)(BsU + tid * LDS_STRIDE);
            dst[0] = d0; dst[1] = d1; dst[2] = d2; dst[3] = d3;
            // prefetch next k-step's chunk (stays inside ws)
            __builtin_prefetch(src + M_DIM * KC, 0, 3);
        }

        __syncthreads();

        // ---- fragments per ISA 7.12.2 layouts ----
        FragBF fa[2];
        #pragma unroll
        for (int si = 0; si < 2; ++si) {
            const unsigned short* arow =
                AsU + (wr * 32 + si * 16 + tl) * LDS_STRIDE;
            fa[si].h[0] = *(const v8bf*)(arow + th * 8);       // K 0..7 / 8..15
            fa[si].h[1] = *(const v8bf*)(arow + 16 + th * 8);  // K 16..23 / 24..31
        }

        #pragma unroll
        for (int sj = 0; sj < 8; ++sj) {
            FragBF fb;
            const unsigned short* brow =
                BsU + (wc * 128 + sj * 16 + tl) * LDS_STRIDE + th * 8;
            fb.h[0] = *(const v8bf*)(brow);
            fb.h[1] = *(const v8bf*)(brow + 16);
            acc[0][sj] = __builtin_amdgcn_wmma_f32_16x16x32_bf16(
                false, fa[0].v, false, fb.v, (short)0, acc[0][sj], false, false);
            acc[1][sj] = __builtin_amdgcn_wmma_f32_16x16x32_bf16(
                false, fa[1].v, false, fb.v, (short)0, acc[1][sj], false, false);
        }

        __syncthreads();
    }

    // ---- epilogue: Abuf[r][m], dims divide exactly, no guards ----
    #pragma unroll
    for (int si = 0; si < 2; ++si) {
        #pragma unroll
        for (int sj = 0; sj < 8; ++sj) {
            const int row0 = r0 + wr * 32 + si * 16 + th * 8;
            const int col  = m0 + wc * 128 + sj * 16 + tl;
            #pragma unroll
            for (int rr = 0; rr < 8; ++rr)
                Abuf[(size_t)(row0 + rr) * M_DIM + col] = acc[si][sj][rr];
        }
    }
}

// --------------------------- stage 2: tiny contraction ----------------------
// out[b,o] = sum_{m,kk} Abuf[b*27+kk][m] * (v-w)[m,kk,o]
__global__ void __launch_bounds__(256)
stage2_reduce_kernel(const float* __restrict__ Abuf,
                     const float* __restrict__ v,
                     const float* __restrict__ w,
                     float* __restrict__ out) {
    const int b   = blockIdx.x;
    const int tid = threadIdx.x;

    float acc[10];
    #pragma unroll
    for (int o = 0; o < 10; ++o) acc[o] = 0.0f;

    for (int idx = tid; idx < M_DIM * 27; idx += 256) {
        int m  = idx / 27;
        int kk = idx - m * 27;
        float a = Abuf[(size_t)(b * 27 + kk) * M_DIM + m];
        const float* pv = v + (m * 27 + kk) * 10;
        const float* pw = w + (m * 27 + kk) * 10;
        #pragma unroll
        for (int o = 0; o < 10; ++o)
            acc[o] = fmaf(a, pv[o] - pw[o], acc[o]);
    }

    __shared__ float red[256 * 10];
    #pragma unroll
    for (int o = 0; o < 10; ++o) red[o * 256 + tid] = acc[o];
    __syncthreads();

    for (int s = 128; s > 0; s >>= 1) {
        if (tid < s) {
            #pragma unroll
            for (int o = 0; o < 10; ++o)
                red[o * 256 + tid] += red[o * 256 + tid + s];
        }
        __syncthreads();
    }
    if (tid < 10) out[b * 10 + tid] = red[tid * 256];
}

// --------------------------- launcher ---------------------------------------
extern "C" void kernel_launch(void* const* d_in, const int* in_sizes, int n_in,
                              void* d_out, int out_size, void* d_ws, size_t ws_size,
                              hipStream_t stream) {
    (void)in_sizes; (void)n_in; (void)out_size; (void)ws_size;

    const float* x = (const float*)d_in[0];   // (512,3,64,64)
    const float* G = (const float*)d_in[1];   // (512,3844)
    const float* v = (const float*)d_in[2];   // (512,27,10)
    const float* w = (const float*)d_in[3];   // (512,27,10)
    float*     out = (float*)d_out;           // (512,10)

    // workspace layout: Gt (bf16, 3.97 MB) | Abuf (f32, 28.3 MB)
    const size_t gt_bytes = (size_t)KITERS * M_DIM * KC * sizeof(unsigned short);
    unsigned short* Gt   = (unsigned short*)d_ws;
    float*          Abuf = (float*)((char*)d_ws + gt_bytes);

    // 1) transpose+convert G into K-blocked bf16
    {
        int total  = KITERS * M_DIM * KC;
        int blocks = (total + 255) / 256;
        prep_gt_kernel<<<blocks, 256, 0, stream>>>(G, Gt);
    }

    // 2) WMMA GEMM: U(13824 x 3844) * G^T(3844 x 512) -> Abuf
    {
        dim3 grid(M_DIM / 256, R_DIM / 128);  // (2, 108)
        unfold_gemm_wmma_kernel<<<grid, 256, 0, stream>>>(x, Gt, Abuf);
    }

    // 3) reduce with (v - w) -> out
    stage2_reduce_kernel<<<512, 256, 0, stream>>>(Abuf, v, w, out);
}
